// MultiheadAttention_62577673503843
// MI455X (gfx1250) — matrix operand order repaired and checked by reading, not compiled, not run on hardware
//
#include <hip/hip_runtime.h>
#include <math.h>

// Problem dims (fixed by the reference)
#define TT   512      // sequence length T
#define BB   8        // batch
#define EE   768      // embed dim
#define HH   32       // heads
#define HD   24       // head dim
#define BH   256      // B*H
#define HDP  32       // head dim padded to 32 for WMMA K tiles

typedef __attribute__((ext_vector_type(16))) __bf16 v16bf;
typedef __attribute__((ext_vector_type(8)))  __bf16 v8bf;
typedef __attribute__((ext_vector_type(8)))  float  v8f;

__device__ __forceinline__ unsigned short f2bfu(float f) {
  unsigned u = __builtin_bit_cast(unsigned, f);
  unsigned r = u + 0x7FFFu + ((u >> 16) & 1u);   // round-to-nearest-even
  return (unsigned short)(r >> 16);
}
__device__ __forceinline__ unsigned pk2bf(float lo, float hi) {
  return (unsigned)f2bfu(lo) | ((unsigned)f2bfu(hi) << 16);
}
// 16-byte vector load of 8 bf16 (pointer must be 16B aligned)
__device__ __forceinline__ v8bf ld8(const unsigned short* p) {
  return __builtin_bit_cast(v8bf, *(const uint4*)p);
}
__device__ __forceinline__ v16bf cat16(v8bf a, v8bf b) {
  return __builtin_shufflevector(a, b, 0,1,2,3,4,5,6,7,8,9,10,11,12,13,14,15);
}
// xor-butterfly max over the 16 lanes of each half (ds_swizzle group-of-32,
// and_mask=0x1f, xor masks 1/2/4/8 -> stays inside each 16-lane half)
__device__ __forceinline__ float xmax16(float v) {
  int x;
  x = __builtin_amdgcn_ds_swizzle(__builtin_bit_cast(int, v), 0x041F);
  v = fmaxf(v, __builtin_bit_cast(float, x));
  x = __builtin_amdgcn_ds_swizzle(__builtin_bit_cast(int, v), 0x081F);
  v = fmaxf(v, __builtin_bit_cast(float, x));
  x = __builtin_amdgcn_ds_swizzle(__builtin_bit_cast(int, v), 0x101F);
  v = fmaxf(v, __builtin_bit_cast(float, x));
  x = __builtin_amdgcn_ds_swizzle(__builtin_bit_cast(int, v), 0x201F);
  v = fmaxf(v, __builtin_bit_cast(float, x));
  return v;
}

// ---------------------------------------------------------------------------
__global__ __launch_bounds__(256) void zero_kernel(uint4* __restrict__ p, long n16) {
  long i = (long)blockIdx.x * blockDim.x + threadIdx.x;
  if (i < n16) p[i] = make_uint4(0u, 0u, 0u, 0u);
}

// ---------------------------------------------------------------------------
// One-time fp32 -> bf16 conversion (4 elements per thread).
// ---------------------------------------------------------------------------
__global__ __launch_bounds__(256) void cvt_bf16_kernel(const float* __restrict__ s,
                                                       unsigned short* __restrict__ d,
                                                       int n4) {
  int i = blockIdx.x * 256 + threadIdx.x;
  if (i < n4) {
    float4 v = ((const float4*)s)[i];
    uint2 o;
    o.x = pk2bf(v.x, v.y);
    o.y = pk2bf(v.z, v.w);
    ((uint2*)d)[i] = o;
  }
}

// ---------------------------------------------------------------------------
// Y = (X @ W^T + bias) * scale, X/W already bf16. One wave = one 16x16 tile,
// K-loop of 32 with v_wmma_f32_16x16x32_bf16, b128 fragment loads.
// MODE 0: f32 row-major out (final output)
// MODE 1: bf16 padded per-head layout (bh, t, 32)  -> Q (scaled) / K
// MODE 2: bf16 transposed layout      (bh, d32, T) -> V^T
// ---------------------------------------------------------------------------
template <int MODE>
__global__ __launch_bounds__(256) void proj_kernel(
    const unsigned short* __restrict__ Xbf, const unsigned short* __restrict__ Wbf,
    const float* __restrict__ bias, float scale,
    unsigned short* __restrict__ obf, float* __restrict__ of32) {
  const int wave = threadIdx.x >> 5;
  const int lane = threadIdx.x & 31;
  const int gid  = blockIdx.x * 8 + wave;       // tile id 0..12287
  const int tn   = gid / 48;                    // row-tile 0..255
  const int tc   = gid - tn * 48;               // col-tile 0..47
  const int half = lane >> 4;
  const int ln   = lane & 15;
  const int n0   = tn * 16, c0 = tc * 16;

  // A: K chunks {0-7,16-23} (lanes 0-15) vs {8-15,24-31} (lanes 16-31)
  const unsigned short* Xr = Xbf + (size_t)(n0 + ln) * EE + (half ? 8 : 0);
  // B: K chunks {0-7,16-23} (lanes 0-15) vs {8-15,24-31} (lanes 16-31), col = row of W (W^T)
  const unsigned short* Wr = Wbf + (size_t)(c0 + ln) * EE + (half ? 8 : 0);

  v8f acc = {};
  for (int k = 0; k < EE; k += 32) {
    v16bf a = cat16(ld8(Xr + k), ld8(Xr + k + 16));
    v16bf b = cat16(ld8(Wr + k), ld8(Wr + k + 16));
    acc = __builtin_amdgcn_wmma_f32_16x16x32_bf16(false, a, false, b,
                                                  (short)0, acc, false, false);
  }

  const int   col = c0 + ln;
  const float bv  = bias[col];
  #pragma unroll
  for (int r = 0; r < 8; ++r) {
    const int   row = n0 + r + half * 8;        // C layout: row = r + 8*half
    const float y   = (acc[r] + bv) * scale;
    if (MODE == 0) {
      of32[(size_t)row * EE + col] = y;
    } else {
      const int t = row >> 3, b = row & 7;      // row = t*B + b (B = 8)
      const int h = col / HD, d = col - h * HD;
      const int bh = b * HH + h;
      if (MODE == 1) obf[((size_t)(bh * TT + t)) * HDP + d] = f2bfu(y);
      else           obf[((size_t)(bh * HDP + d)) * TT + t] = f2bfu(y);
    }
  }
}

// ---------------------------------------------------------------------------
// Flash-style attention: one wave = one (bh, 16-query-row) tile.
// Per 32-key step: 2 score WMMAs (+bias, +mask), online softmax where the
// row-max uses ds_swizzle xor-butterflies and the row-sum is a WMMA against a
// constant all-ones B matrix (C-layout hands each lane its own row's sum).
// P goes C-layout -> LDS -> A-layout for the 2 PV WMMAs.
// ---------------------------------------------------------------------------
__global__ __launch_bounds__(256) void attn_kernel(
    const unsigned short* __restrict__ qpad,
    const unsigned short* __restrict__ kpad,
    const unsigned short* __restrict__ vT,
    const float* __restrict__ attn_bias,
    const unsigned char* __restrict__ kmask,
    unsigned short* __restrict__ attnO) {
  __shared__ alignas(16) unsigned short ldsP[8][16 * 32];   // 1 KB per wave
  const int wave = threadIdx.x >> 5;
  const int lane = threadIdx.x & 31;
  const int gid  = blockIdx.x * 8 + wave;       // 0..8191
  const int bh   = gid >> 5;                    // 0..255
  const int qt   = gid & 31;                    // query tile 0..31
  const int b    = bh >> 5;
  const int h    = bh & 31;
  const int half = lane >> 4;
  const int ln   = lane & 15;

  // constant all-ones B fragment (bf16 1.0 = 0x3F80) for row sums
  v16bf ones;
  #pragma unroll
  for (int j = 0; j < 16; ++j) ones[j] = __builtin_bit_cast(__bf16, (unsigned short)0x3F80);

  // Q A-fragment (rows qt*16..+15, K = padded head dim 0..31)
  const unsigned short* qr =
      qpad + ((size_t)(bh * TT + qt * 16 + ln)) * HDP + (half ? 8 : 0);
  const v16bf aq = cat16(ld8(qr), ld8(qr + 16));

  float m[8], l[8];
  v8f o0 = {}, o1 = {};
  #pragma unroll
  for (int r = 0; r < 8; ++r) { m[r] = -1e30f; l[r] = 0.0f; }

  const float*         biasB = attn_bias + (size_t)bh * TT * TT;
  const unsigned char* mrow  = kmask + b * TT;

  for (int kt2 = 0; kt2 < TT / 32; ++kt2) {
    const int k0 = kt2 * 32;
    float p[2][8];

    #pragma unroll
    for (int t = 0; t < 2; ++t) {
      const int kr = k0 + t * 16 + ln;          // this lane's key column
      const unsigned short* kp =
          kpad + ((size_t)(bh * TT + kr)) * HDP + (half ? 8 : 0);
      const v16bf bk = cat16(ld8(kp), ld8(kp + 16));
      v8f c = {};
      c = __builtin_amdgcn_wmma_f32_16x16x32_bf16(false, aq, false, bk,
                                                  (short)0, c, false, false);
      const bool dead = (mrow[kr] != 0);
      #pragma unroll
      for (int r = 0; r < 8; ++r) {
        const int row = qt * 16 + r + half * 8;
        const float s = c[r] + biasB[(size_t)row * TT + kr];
        p[t][r] = dead ? -1e30f : s;
      }
    }

    // online rescale: row-max across the 32 columns via xor-butterflies
    float alpha[8];
    #pragma unroll
    for (int r = 0; r < 8; ++r) {
      const float rm = xmax16(fmaxf(p[0][r], p[1][r]));
      const float mn = fmaxf(m[r], rm);
      alpha[r] = __expf(m[r] - mn);
      m[r]     = mn;
      p[0][r]  = __expf(p[0][r] - mn);
      p[1][r]  = __expf(p[1][r] - mn);
    }
    #pragma unroll
    for (int r = 0; r < 8; ++r) { o0[r] *= alpha[r]; o1[r] *= alpha[r]; }

    // stage P: C-layout -> LDS -> A-layout (private per-wave region; LDS in-order)
    unsigned short* lp = ldsP[wave];
    #pragma unroll
    for (int r = 0; r < 8; ++r) {
      const int row = r + half * 8;
      lp[row * 32 + ln]      = f2bfu(p[0][r]);
      lp[row * 32 + 16 + ln] = f2bfu(p[1][r]);
    }
    asm volatile("s_wait_dscnt 0" ::: "memory");
    const unsigned short* pr = lp + ln * 32 + (half ? 8 : 0);
    const v16bf pa = cat16(ld8(pr), ld8(pr + 16));

    // row sums: P x ones -> every column of row m holds sum_k P[m,k]
    {
      v8f rs = {};
      rs = __builtin_amdgcn_wmma_f32_16x16x32_bf16(false, pa, false, ones,
                                                   (short)0, rs, false, false);
      #pragma unroll
      for (int r = 0; r < 8; ++r) l[r] = l[r] * alpha[r] + rs[r];
    }

    // PV: B from V^T (K chunks {0-7,16-23} / {8-15,24-31}), two 16-col output tiles
    {
      const unsigned short* vp0 =
          vT + ((size_t)(bh * HDP + ln)) * TT + k0 + (half ? 8 : 0);
      const v16bf vb0 = cat16(ld8(vp0), ld8(vp0 + 16));
      o0 = __builtin_amdgcn_wmma_f32_16x16x32_bf16(false, pa, false, vb0,
                                                   (short)0, o0, false, false);
      const unsigned short* vp1 =
          vT + ((size_t)(bh * HDP + 16 + ln)) * TT + k0 + (half ? 8 : 0);
      const v16bf vb1 = cat16(ld8(vp1), ld8(vp1 + 16));
      o1 = __builtin_amdgcn_wmma_f32_16x16x32_bf16(false, pa, false, vb1,
                                                   (short)0, o1, false, false);
    }
  }

  // epilogue: normalize and scatter into (T, B, E) as bf16
  #pragma unroll
  for (int r = 0; r < 8; ++r) {
    const int   trow = qt * 16 + r + half * 8;
    const float inv  = 1.0f / l[r];
    attnO[((size_t)trow * BB + b) * EE + h * HD + ln] = f2bfu(o0[r] * inv);
    const int d1 = 16 + ln;
    if (d1 < HD)
      attnO[((size_t)trow * BB + b) * EE + h * HD + d1] = f2bfu(o1[r] * inv);
  }
}

// ---------------------------------------------------------------------------
extern "C" void kernel_launch(void* const* d_in, const int* in_sizes, int n_in,
                              void* d_out, int out_size, void* d_ws, size_t ws_size,
                              hipStream_t stream) {
  const float*         query = (const float*)d_in[0];
  const float*         abias = (const float*)d_in[1];
  const unsigned char* kmask = (const unsigned char*)d_in[2];
  const float* Wq = (const float*)d_in[3];
  const float* bq = (const float*)d_in[4];
  const float* Wk = (const float*)d_in[5];
  const float* bk = (const float*)d_in[6];
  const float* Wv = (const float*)d_in[7];
  const float* bv = (const float*)d_in[8];
  const float* Wo = (const float*)d_in[9];
  const float* bo = (const float*)d_in[10];
  float* out = (float*)d_out;

  // workspace layout (ushort elements)
  const size_t qkvElems = (size_t)BH * TT * HDP;   // 4,194,304 each
  const size_t xElems   = (size_t)TT * BB * EE;    // 3,145,728
  const size_t wElems   = (size_t)EE * EE;         //   589,824
  unsigned short* qpad  = (unsigned short*)d_ws;
  unsigned short* kpad  = qpad + qkvElems;
  unsigned short* vTp   = kpad + qkvElems;
  unsigned short* attnO = vTp + qkvElems;
  unsigned short* Xbf   = attnO + xElems;
  unsigned short* Wqbf  = Xbf + xElems;
  unsigned short* Wkbf  = Wqbf + wElems;
  unsigned short* Wvbf  = Wkbf + wElems;
  unsigned short* Wobf  = Wvbf + wElems;

  // 1) zero the padded bf16 q/k/vT buffers (padding lanes must stay 0)
  const long n16 = (long)(3 * qkvElems * sizeof(unsigned short) / 16);
  zero_kernel<<<(unsigned)((n16 + 255) / 256), 256, 0, stream>>>((uint4*)d_ws, n16);

  // 2) one-time fp32 -> bf16 conversions
  const int xq4 = (int)(xElems / 4);   // 786,432 -> 3072 blocks
  const int w4  = (int)(wElems / 4);   // 147,456 ->  576 blocks
  cvt_bf16_kernel<<<(xq4 + 255) / 256, 256, 0, stream>>>(query, Xbf, xq4);
  cvt_bf16_kernel<<<(w4 + 255) / 256, 256, 0, stream>>>(Wq, Wqbf, w4);
  cvt_bf16_kernel<<<(w4 + 255) / 256, 256, 0, stream>>>(Wk, Wkbf, w4);
  cvt_bf16_kernel<<<(w4 + 255) / 256, 256, 0, stream>>>(Wv, Wvbf, w4);
  cvt_bf16_kernel<<<(w4 + 255) / 256, 256, 0, stream>>>(Wo, Wobf, w4);

  // 3) projections: 12288 tiles, 8 waves/block -> 1536 blocks of 256
  const float scaleQ = 1.0f / sqrtf((float)HD);
  proj_kernel<1><<<1536, 256, 0, stream>>>(Xbf, Wqbf, bq, scaleQ, qpad, nullptr);
  proj_kernel<1><<<1536, 256, 0, stream>>>(Xbf, Wkbf, bk, 1.0f,   kpad, nullptr);
  proj_kernel<2><<<1536, 256, 0, stream>>>(Xbf, Wvbf, bv, 1.0f,   vTp,  nullptr);

  // 4) attention: 8192 waves -> 1024 blocks of 256
  attn_kernel<<<1024, 256, 0, stream>>>(qpad, kpad, vTp, abias, kmask, attnO);

  // 5) output projection straight into d_out
  proj_kernel<0><<<1536, 256, 0, stream>>>(attnO, Wobf, bo, 1.0f, nullptr, out);
}
